// ple_gru_30039001268308
// MI455X (gfx1250) — matrix operand order repaired and checked by reading, not compiled, not run on hardware
//
#include <hip/hip_runtime.h>

#define BB 4096
#define TT 64
#define DD 16
#define HH 32
#define WAVES 4

typedef __attribute__((ext_vector_type(16))) __bf16 v16bf;
typedef __attribute__((ext_vector_type(8)))  float  v8f;

union Frag { v16bf v; unsigned int u[8]; };

__device__ __forceinline__ unsigned int f2bf(float f) {
  union { float f; unsigned int u; } c; c.f = f;
  unsigned int r = c.u + 0x7FFFu + ((c.u >> 16) & 1u);   // round-to-nearest-even
  return r >> 16;
}

// gfx1250 v_tanh_f32 trans op (probe-confirmed present); branchless fallback.
__device__ __forceinline__ float fast_tanh(float x) {
#if __has_builtin(__builtin_amdgcn_tanhf)
  return __builtin_amdgcn_tanhf(x);
#else
  const float e = __builtin_amdgcn_exp2f(2.885390081777927f * x);  // e^(2x)
  return 1.0f - 2.0f * __builtin_amdgcn_rcpf(e + 1.0f);
#endif
}
__device__ __forceinline__ float sigm(float x) {
#if __has_builtin(__builtin_amdgcn_tanhf)
  return fmaf(0.5f, __builtin_amdgcn_tanhf(0.5f * x), 0.5f);
#else
  return __builtin_amdgcn_rcpf(1.0f + __builtin_amdgcn_exp2f(-1.4426950408889634f * x));
#endif
}

// Wave-private LDS ordering fence: LDS ops of one wave complete in order
// (DScnt), we only need to stop compiler reordering across the exchange.
__device__ __forceinline__ void wave_fence() { __builtin_amdgcn_wave_barrier(); }

// -------------------------------------------------------------------------
// Batched GRU layer: gridDim.y selects one of 3 independent stacks.
// Each wave owns 16 batch rows; per timestep the 16x96 projection is done
// with 12 v_wmma_f32_16x16x32_bf16 (input-K zero-padded to 32 so the x-side
// GEMM fuses into the same accumulators).  Weights live in VGPRs as
// B-fragments; h is mirrored in wave-private LDS (bf16) for the A-fragment
// relayout; elementwise gate math stays f32 (v_tanh_f32).
// -------------------------------------------------------------------------
struct Gru3 {
  const float* xs[3];
  const float* wih[3];
  const float* whh[3];
  const float* bih[3];
  const float* bhh[3];
  float*       out[3];
};

__global__ __launch_bounds__(WAVES * 32) void gru_layer_wmma(Gru3 a, int Din)
{
  __shared__ alignas(16) unsigned short hbuf[WAVES][16][32];
  __shared__ alignas(16) unsigned short xbuf[WAVES][16][32];

  const int s = blockIdx.y;
  const float* __restrict__ xseq = a.xs[s];
  const float* __restrict__ Wih  = a.wih[s];
  const float* __restrict__ Whh  = a.whh[s];
  const float* __restrict__ bih  = a.bih[s];
  const float* __restrict__ bhh  = a.bhh[s];
  float* __restrict__ hseq       = a.out[s];

  const int tid  = threadIdx.x;
  const int wave = tid >> 5;
  const int lane = tid & 31;
  const int nl   = lane & 15;   // column within tile (B/C layout), row for A loads
  const int kh   = lane >> 4;   // K-half selector
  const int b0   = blockIdx.x * (WAVES * 16) + wave * 16;

  // ---- B-fragments: tile j covers output columns 16j..16j+15.
  // B(k, n) = W[16j + n, k]; lane n reads 16 consecutive floats of row (16j+n),
  // starting at column 16*kh. Zero-pad K beyond Din for input-side weights.
  Frag Bih[6], Bhh[6];
#pragma unroll
  for (int j = 0; j < 6; ++j) {
    const int row = j * 16 + nl;
#pragma unroll
    for (int e2 = 0; e2 < 8; ++e2) {
      const int k0 = kh * 16 + e2 * 2;
      const int k1 = k0 + 1;
      unsigned lo = (k0 < Din) ? f2bf(Wih[row * Din + k0]) : 0u;
      unsigned hi = (k1 < Din) ? f2bf(Wih[row * Din + k1]) : 0u;
      Bih[j].u[e2] = lo | (hi << 16);
      Bhh[j].u[e2] = f2bf(Whh[row * HH + k0]) | (f2bf(Whh[row * HH + k1]) << 16);
    }
  }

  float bihs[6], bhhs[6];
#pragma unroll
  for (int j = 0; j < 6; ++j) { bihs[j] = bih[j * 16 + nl]; bhhs[j] = bhh[j * 16 + nl]; }

  // zero-init h mirror and x staging (upper half of xbuf stays 0 when Din==16)
  for (int i = lane; i < 16 * 32; i += 32) {
    (&hbuf[wave][0][0])[i] = 0;
    (&xbuf[wave][0][0])[i] = 0;
  }
  wave_fence();

  v8f h0, h1;
#pragma unroll
  for (int i = 0; i < 8; ++i) { h0[i] = 0.f; h1[i] = 0.f; }

  const int xr_row = lane >> 1;   // staging: each lane handles half a row

  for (int t = 0; t < TT; ++t) {
    // ---- stage x(t) for this wave's 16 rows into LDS (bf16, vectorized)
    if (Din == 16) {
      const int c0 = (lane & 1) * 8;    // 8 cols per lane
      const float* src = xseq + ((size_t)(b0 + xr_row) * TT + t) * 16 + c0;
      const float4 va = *(const float4*)(src);
      const float4 vb = *(const float4*)(src + 4);
      uint4 pk;
      pk.x = f2bf(va.x) | (f2bf(va.y) << 16);
      pk.y = f2bf(va.z) | (f2bf(va.w) << 16);
      pk.z = f2bf(vb.x) | (f2bf(vb.y) << 16);
      pk.w = f2bf(vb.z) | (f2bf(vb.w) << 16);
      *(uint4*)(&xbuf[wave][xr_row][c0]) = pk;
    } else {                            // Din == 32, 16 cols per lane
      const int c0 = (lane & 1) * 16;
      const float* src = xseq + ((size_t)(b0 + xr_row) * TT + t) * 32 + c0;
#pragma unroll
      for (int half = 0; half < 2; ++half) {
        const float4 va = *(const float4*)(src + half * 8);
        const float4 vb = *(const float4*)(src + half * 8 + 4);
        uint4 pk;
        pk.x = f2bf(va.x) | (f2bf(va.y) << 16);
        pk.y = f2bf(va.z) | (f2bf(va.w) << 16);
        pk.z = f2bf(vb.x) | (f2bf(vb.y) << 16);
        pk.w = f2bf(vb.z) | (f2bf(vb.w) << 16);
        *(uint4*)(&xbuf[wave][xr_row][c0 + half * 8]) = pk;
      }
    }
    wave_fence();

    // ---- A-fragments: lane<16 -> row=lane,  K {0..15}
    //                   lane>=16 -> row=lane-16, K {16..31}
    Frag Ax, Ah;
    {
      const unsigned short* xr = &xbuf[wave][nl][0];
      const unsigned short* hr = &hbuf[wave][nl][0];
      const uint4 xc1 = *(const uint4*)(xr + kh * 16);
      const uint4 xc2 = *(const uint4*)(xr + 8 + kh * 16);
      const uint4 hc1 = *(const uint4*)(hr + kh * 16);
      const uint4 hc2 = *(const uint4*)(hr + 8 + kh * 16);
      Ax.u[0] = xc1.x; Ax.u[1] = xc1.y; Ax.u[2] = xc1.z; Ax.u[3] = xc1.w;
      Ax.u[4] = xc2.x; Ax.u[5] = xc2.y; Ax.u[6] = xc2.z; Ax.u[7] = xc2.w;
      Ah.u[0] = hc1.x; Ah.u[1] = hc1.y; Ah.u[2] = hc1.z; Ah.u[3] = hc1.w;
      Ah.u[4] = hc2.x; Ah.u[5] = hc2.y; Ah.u[6] = hc2.z; Ah.u[7] = hc2.w;
    }
    wave_fence();

#pragma unroll
    for (int g = 0; g < 2; ++g) {     // two 16-column halves of the hidden state
      v8f accr, accz, accxn, acchn;
#pragma unroll
      for (int i = 0; i < 8; ++i) {
        accr[i]  = bihs[g]     + bhhs[g];
        accz[i]  = bihs[2 + g] + bhhs[2 + g];
        accxn[i] = bihs[4 + g];          // n-gate keeps x-side and h-side apart
        acchn[i] = bhhs[4 + g];
      }
      // grouped by A operand for scheduler locality
      accr  = __builtin_amdgcn_wmma_f32_16x16x32_bf16(false, Ax.v, false, Bih[g].v,     (short)0, accr,  false, false);
      accz  = __builtin_amdgcn_wmma_f32_16x16x32_bf16(false, Ax.v, false, Bih[2 + g].v, (short)0, accz,  false, false);
      accxn = __builtin_amdgcn_wmma_f32_16x16x32_bf16(false, Ax.v, false, Bih[4 + g].v, (short)0, accxn, false, false);
      accr  = __builtin_amdgcn_wmma_f32_16x16x32_bf16(false, Ah.v, false, Bhh[g].v,     (short)0, accr,  false, false);
      accz  = __builtin_amdgcn_wmma_f32_16x16x32_bf16(false, Ah.v, false, Bhh[2 + g].v, (short)0, accz,  false, false);
      acchn = __builtin_amdgcn_wmma_f32_16x16x32_bf16(false, Ah.v, false, Bhh[4 + g].v, (short)0, acchn, false, false);

      const v8f hv = g ? h1 : h0;
      v8f hn;
#pragma unroll
      for (int i = 0; i < 8; ++i) {
        const float rr = sigm(accr[i]);
        const float zz = sigm(accz[i]);
        const float nn = fast_tanh(accxn[i] + rr * acchn[i]);
        const float nh = (1.f - zz) * nn + zz * hv[i];
        hn[i] = nh;
        const int m = kh * 8 + i;
        hbuf[wave][m][g * 16 + nl] = (unsigned short)f2bf(nh);
        hseq[((size_t)(b0 + m) * TT + t) * HH + g * 16 + nl] = nh;
      }
      if (g) h1 = hn; else h0 = hn;
    }
    wave_fence();
  }
}

// -------------------------------------------------------------------------
// Tiny 2-layer gate GRU (hidden KH = 2 or 3), one thread per batch row,
// NSTK stacks batched on gridDim.y; softmax of final hidden state.
// -------------------------------------------------------------------------
template <int KH, int NSTK>
struct GateArgs { const float* w[NSTK][8]; float* v[NSTK]; };

template <int KH, int NSTK>
__global__ __launch_bounds__(256) void gate_gru(const float* __restrict__ x,
                                               GateArgs<KH, NSTK> ga)
{
  constexpr int G = 3 * KH;
  __shared__ float sWih0[G * DD], sWhh0[G * KH], sb0[2 * G];
  __shared__ float sWih1[G * KH], sWhh1[G * KH], sb1[2 * G];
  const int s = blockIdx.y;
  const float* Wih0 = ga.w[s][0]; const float* Whh0 = ga.w[s][1];
  const float* bih0 = ga.w[s][2]; const float* bhh0 = ga.w[s][3];
  const float* Wih1 = ga.w[s][4]; const float* Whh1 = ga.w[s][5];
  const float* bih1 = ga.w[s][6]; const float* bhh1 = ga.w[s][7];
  float* vout = ga.v[s];

  const int tid = threadIdx.x;
  for (int i = tid; i < G * DD; i += 256) sWih0[i] = Wih0[i];
  for (int i = tid; i < G * KH; i += 256) {
    sWhh0[i] = Whh0[i]; sWih1[i] = Wih1[i]; sWhh1[i] = Whh1[i];
  }
  for (int i = tid; i < G; i += 256) {
    sb0[i] = bih0[i]; sb0[G + i] = bhh0[i];
    sb1[i] = bih1[i]; sb1[G + i] = bhh1[i];
  }
  __syncthreads();

  const int b = blockIdx.x * 256 + tid;
  float h0[KH], h1[KH];
#pragma unroll
  for (int j = 0; j < KH; ++j) { h0[j] = 0.f; h1[j] = 0.f; }

  for (int t = 0; t < TT; ++t) {
    float xr[DD];
    {
      const float4* xp = (const float4*)(x + ((size_t)b * TT + t) * DD);
#pragma unroll
      for (int q = 0; q < DD / 4; ++q) {
        const float4 v4 = xp[q];
        xr[4 * q + 0] = v4.x; xr[4 * q + 1] = v4.y;
        xr[4 * q + 2] = v4.z; xr[4 * q + 3] = v4.w;
      }
    }

    float a[G], p[G];
#pragma unroll
    for (int g = 0; g < G; ++g) {
      float acc = sb0[g];
#pragma unroll
      for (int d = 0; d < DD; ++d) acc += sWih0[g * DD + d] * xr[d];
      a[g] = acc;
      float q = sb0[G + g];
#pragma unroll
      for (int j = 0; j < KH; ++j) q += sWhh0[g * KH + j] * h0[j];
      p[g] = q;
    }
#pragma unroll
    for (int j = 0; j < KH; ++j) {
      const float r = sigm(a[j] + p[j]);
      const float z = sigm(a[KH + j] + p[KH + j]);
      const float n = fast_tanh(a[2 * KH + j] + r * p[2 * KH + j]);
      h0[j] = (1.f - z) * n + z * h0[j];
    }

    float a1[G], p1[G];
#pragma unroll
    for (int g = 0; g < G; ++g) {
      float acc = sb1[g];
#pragma unroll
      for (int j = 0; j < KH; ++j) acc += sWih1[g * KH + j] * h0[j];
      a1[g] = acc;
      float q = sb1[G + g];
#pragma unroll
      for (int j = 0; j < KH; ++j) q += sWhh1[g * KH + j] * h1[j];
      p1[g] = q;
    }
#pragma unroll
    for (int j = 0; j < KH; ++j) {
      const float r = sigm(a1[j] + p1[j]);
      const float z = sigm(a1[KH + j] + p1[KH + j]);
      const float n = fast_tanh(a1[2 * KH + j] + r * p1[2 * KH + j]);
      h1[j] = (1.f - z) * n + z * h1[j];
    }
  }
  // softmax over the KH experts
  float m = h1[0];
#pragma unroll
  for (int j = 1; j < KH; ++j) m = fmaxf(m, h1[j]);
  float sum = 0.f, e[KH];
#pragma unroll
  for (int j = 0; j < KH; ++j) { e[j] = __expf(h1[j] - m); sum += e[j]; }
  const float inv = 1.f / sum;
#pragma unroll
  for (int j = 0; j < KH; ++j) vout[b * KH + j] = e[j] * inv;
}

// out[b,t,h] = sum_k v[b,k] * expert_k[b,t,h]; three combines batched on y.
struct Comb3 {
  const float* e0[3]; const float* e1[3]; const float* e2[3];
  const float* v[3];  int K[3];  float* out[3];
};

__global__ __launch_bounds__(256) void combine_k(Comb3 c)
{
  const int s = blockIdx.y;
  const size_t idx = (size_t)blockIdx.x * 256 + threadIdx.x;
  const int b = (int)(idx / (TT * HH));
  const int K = c.K[s];
  float acc = c.v[s][b * K + 0] * c.e0[s][idx] + c.v[s][b * K + 1] * c.e1[s][idx];
  if (K == 3) acc += c.v[s][b * K + 2] * c.e2[s][idx];
  c.out[s][idx] = acc;
}

// leaky_relu(t[:, -1]) @ W^T + b, then eval-mode BatchNorm with running (0,1)
__global__ __launch_bounds__(256) void heads_k(
    const float* __restrict__ t0, const float* __restrict__ t1,
    const float* __restrict__ W0, const float* __restrict__ b0v,
    const float* __restrict__ g0, const float* __restrict__ be0,
    const float* __restrict__ W1, const float* __restrict__ b1v,
    const float* __restrict__ g1, const float* __restrict__ be1,
    float* __restrict__ out)
{
  const int b = blockIdx.x * 256 + threadIdx.x;
  if (b >= BB) return;
  const float bn = 0.99999500003749981f;   // 1/sqrt(1 + 1e-5)
#pragma unroll
  for (int i = 0; i < 2; ++i) {
    const float* seq = i ? t1 : t0;
    const float* W   = i ? W1 : W0;
    const float bb   = i ? b1v[0] : b0v[0];
    const float gg   = i ? g1[0]  : g0[0];
    const float be   = i ? be1[0] : be0[0];
    float acc = 0.f;
#pragma unroll
    for (int h = 0; h < HH; ++h) {
      float vv = seq[((size_t)b * TT + (TT - 1)) * HH + h];
      vv = vv > 0.f ? vv : 0.01f * vv;
      acc += vv * W[h];
    }
    out[b * 2 + i] = (acc + bb) * bn * gg + be;
  }
}

// -------------------------------------------------------------------------
extern "C" void kernel_launch(void* const* d_in, const int* in_sizes, int n_in,
                              void* d_out, int out_size, void* d_ws, size_t ws_size,
                              hipStream_t stream)
{
  (void)in_sizes; (void)n_in; (void)out_size; (void)ws_size;

  const float* x = (const float*)d_in[0];
  int p = 1;
  struct GruP { const float *Wih, *Whh, *bih, *bhh; };
  auto nextG = [&](GruP& g) {
    g.Wih = (const float*)d_in[p++]; g.Whh = (const float*)d_in[p++];
    g.bih = (const float*)d_in[p++]; g.bhh = (const float*)d_in[p++];
  };
  GruP se[2][2], te[2][2][2], sg[2][2], tg[2][2][2];
  for (int l = 0; l < 2; ++l) {
    nextG(se[l][0]); nextG(se[l][1]);
    for (int k = 0; k < 2; ++k) { nextG(te[l][k][0]); nextG(te[l][k][1]); }
    nextG(sg[l][0]); nextG(sg[l][1]);
    for (int k = 0; k < 2; ++k) { nextG(tg[l][k][0]); nextG(tg[l][k][1]); }
  }
  const float *hW[2], *hb[2], *hg[2], *hbe[2];
  for (int i = 0; i < 2; ++i) {
    hW[i]  = (const float*)d_in[p++]; hb[i]  = (const float*)d_in[p++];
    hg[i]  = (const float*)d_in[p++]; hbe[i] = (const float*)d_in[p++];
  }

  float* ws = (float*)d_ws;
  const size_t S = (size_t)BB * TT * HH;
  float* tmp[3] = { ws, ws + S, ws + 2 * S };     // layer-0 intermediates
  float* d_sv  = ws + 3 * S;
  float* d_tv0 = ws + 4 * S;
  float* d_tv1 = ws + 5 * S;
  float* d_sh  = ws + 6 * S;      // shared path
  float* d_t0  = ws + 7 * S;      // task paths
  float* d_t1  = ws + 8 * S;
  float* gv_s  = ws + 9 * S;      // (B,3)
  float* gv_t0 = gv_s  + (size_t)BB * 3;
  float* gv_t1 = gv_t0 + (size_t)BB * 2;

  const dim3 gGru(BB / (WAVES * 16), 3), bGru(WAVES * 32);
  const dim3 gComb((unsigned)(S / 256), 3);

  for (int l = 0; l < 2; ++l) {
    const float* inS  = l ? d_sh : x;
    const float* inT0 = l ? d_t0 : x;
    const float* inT1 = l ? d_t1 : x;
    const int Din = l ? HH : DD;

    // layer-0 of all 3 expert stacks in one launch, then layer-1
    Gru3 a0, a1;
    GruP* g0s[3] = { &se[l][0], &te[l][0][0], &te[l][1][0] };
    GruP* g1s[3] = { &se[l][1], &te[l][0][1], &te[l][1][1] };
    const float* ins[3] = { inS, inT0, inT1 };
    float* outs[3] = { d_sv, d_tv0, d_tv1 };
    for (int k = 0; k < 3; ++k) {
      a0.xs[k] = ins[k];  a0.wih[k] = g0s[k]->Wih; a0.whh[k] = g0s[k]->Whh;
      a0.bih[k] = g0s[k]->bih; a0.bhh[k] = g0s[k]->bhh; a0.out[k] = tmp[k];
      a1.xs[k] = tmp[k];  a1.wih[k] = g1s[k]->Wih; a1.whh[k] = g1s[k]->Whh;
      a1.bih[k] = g1s[k]->bih; a1.bhh[k] = g1s[k]->bhh; a1.out[k] = outs[k];
    }
    gru_layer_wmma<<<gGru, bGru, 0, stream>>>(a0, Din);
    gru_layer_wmma<<<gGru, bGru, 0, stream>>>(a1, HH);

    // gates: shared (K=3) single launch; both task gates (K=2) batched
    GateArgs<3, 1> gs;
    gs.w[0][0] = sg[l][0].Wih; gs.w[0][1] = sg[l][0].Whh;
    gs.w[0][2] = sg[l][0].bih; gs.w[0][3] = sg[l][0].bhh;
    gs.w[0][4] = sg[l][1].Wih; gs.w[0][5] = sg[l][1].Whh;
    gs.w[0][6] = sg[l][1].bih; gs.w[0][7] = sg[l][1].bhh;
    gs.v[0] = gv_s;
    gate_gru<3, 1><<<dim3(BB / 256, 1), 256, 0, stream>>>(x, gs);

    GateArgs<2, 2> gt;
    float* gvt[2] = { gv_t0, gv_t1 };
    for (int k = 0; k < 2; ++k) {
      gt.w[k][0] = tg[l][k][0].Wih; gt.w[k][1] = tg[l][k][0].Whh;
      gt.w[k][2] = tg[l][k][0].bih; gt.w[k][3] = tg[l][k][0].bhh;
      gt.w[k][4] = tg[l][k][1].Wih; gt.w[k][5] = tg[l][k][1].Whh;
      gt.w[k][6] = tg[l][k][1].bih; gt.w[k][7] = tg[l][k][1].bhh;
      gt.v[k] = gvt[k];
    }
    gate_gru<2, 2><<<dim3(BB / 256, 2), 256, 0, stream>>>(x, gt);

    // three gated mixtures in one launch
    Comb3 c;
    c.e0[0] = d_sv; c.e1[0] = d_tv0; c.e2[0] = d_tv1; c.v[0] = gv_s;  c.K[0] = 3; c.out[0] = d_sh;
    c.e0[1] = d_sv; c.e1[1] = d_tv0; c.e2[1] = d_tv0; c.v[1] = gv_t0; c.K[1] = 2; c.out[1] = d_t0;
    c.e0[2] = d_sv; c.e1[2] = d_tv1; c.e2[2] = d_tv1; c.v[2] = gv_t1; c.K[2] = 2; c.out[2] = d_t1;
    combine_k<<<gComb, 256, 0, stream>>>(c);
  }

  heads_k<<<BB / 256, 256, 0, stream>>>(d_t0, d_t1,
                                        hW[0], hb[0], hg[0], hbe[0],
                                        hW[1], hb[1], hg[1], hbe[1],
                                        (float*)d_out);
}
